// ModulatedAttentionBlock_78554951843892
// MI455X (gfx1250) — matrix operand order repaired and checked by reading, not compiled, not run on hardware
//
#include <hip/hip_runtime.h>

// ---------------------------------------------------------------------------
// CDNA5 (gfx1250, wave32) modulated-attention block.
// bf16 WMMA f32-accumulate + async-LDS copies (ASYNCcnt) + TDM tensor loads
// (TENSORcnt) for the attention K/V tiles.
// ---------------------------------------------------------------------------

typedef __attribute__((ext_vector_type(16))) __bf16 v16bf;
typedef __attribute__((ext_vector_type(8)))  float  v8f;
typedef __attribute__((ext_vector_type(8)))  unsigned short ushort8;
typedef unsigned int uint4v __attribute__((ext_vector_type(4)));
typedef int          int4v  __attribute__((ext_vector_type(4)));
typedef int          int8v  __attribute__((ext_vector_type(8)));

constexpr int Bc = 2, Lc = 2048, Dc = 2048, Hc = 16, KVHc = 8, Dhc = 128;
constexpr int SROPE = 1024;
constexpr int D3 = 3 * Dc;

union Frag { v16bf v; ushort8 h[2]; };

__device__ __forceinline__ unsigned short f2bf(float f) {
  unsigned int u = __builtin_bit_cast(unsigned int, f);
  u += 0x7FFFu + ((u >> 16) & 1u);              // round-to-nearest-even
  return (unsigned short)(u >> 16);
}

__device__ __forceinline__ v8f vzero8() {
  v8f z;
#pragma unroll
  for (int i = 0; i < 8; ++i) z[i] = 0.0f;
  return z;
}

__device__ __forceinline__ v8f wmma_bf16(Frag a, Frag b, v8f c) {
  return __builtin_amdgcn_wmma_f32_16x16x32_bf16(false, a.v, false, b.v,
                                                 (short)0, c, false, false);
}

// Async global->LDS 16B copy (per-lane), tracked with ASYNCcnt.
__device__ __forceinline__ void async_ld_b128(void* lds, const void* g) {
  unsigned l = (unsigned)(unsigned long long)lds;   // LDS offset = addr[31:0]
  asm volatile("global_load_async_to_lds_b128 %0, %1, off"
               :: "v"(l), "v"(g) : "memory");
}
__device__ __forceinline__ void wait_async0() {
  asm volatile("s_wait_asynccnt 0x0" ::: "memory");
}

// TDM: 2D tile (tile_d1 rows of tile_d0 elements, 2B/elem, row stride stride0
// elements) from global to contiguous LDS. D# per ISA 8.3/8.4.
__device__ __forceinline__ void tdm_load_2d(unsigned lds_addr, const void* gp,
                                            unsigned tile_d0, unsigned tile_d1,
                                            unsigned long long td0,
                                            unsigned long long td1,
                                            unsigned long long stride0) {
  unsigned long long ga = (unsigned long long)gp;
  uint4v g0;
  g0.x = 1u;                                                   // count=1 (valid)
  g0.y = lds_addr;                                             // lds_addr
  g0.z = (unsigned)ga;                                         // global[31:0]
  g0.w = (unsigned)((ga >> 32) & 0x01FFFFFFull) | (2u << 30);  // global hi, type=2
  int8v g1;
  g1[0] = (int)(1u << 16);                                     // data_size=2B
  g1[1] = (int)((unsigned)(td0 & 0xFFFFull) << 16);            // tensor_dim0 lo
  g1[2] = (int)(((td0 >> 16) & 0xFFFFull) | ((td1 & 0xFFFFull) << 16));
  g1[3] = (int)(((td1 >> 16) & 0xFFFFull) | ((unsigned long long)tile_d0 << 16));
  g1[4] = (int)(tile_d1 & 0xFFFFu);                            // tile_dim1, dim2=0
  g1[5] = (int)(stride0 & 0xFFFFFFFFull);                      // dim0 stride lo
  g1[6] = (int)((stride0 >> 32) & 0xFFFFull);                  // dim0 stride hi
  g1[7] = 0;
  int4v gz = {};
#if __clang_major__ >= 23
  int8v gz8 = {};
  __builtin_amdgcn_tensor_load_to_lds(g0, g1, gz, gz, gz8, 0);
#else
  __builtin_amdgcn_tensor_load_to_lds(g0, g1, gz, gz, 0);
#endif
}

// ---------------------------------------------------------------------------
// 1) mod = silu(temb) @ w_mod + b_mod        (B x 3D), fp32
// ---------------------------------------------------------------------------
__global__ __launch_bounds__(256) void mod_kernel(
    const float* __restrict__ temb, const float* __restrict__ w_mod,
    const float* __restrict__ b_mod, float* __restrict__ mod) {
  int col = blockIdx.x * 256 + threadIdx.x;
  int b   = blockIdx.y;
  if (col >= D3) return;
  const float* t = temb + (size_t)b * Dc;
  float acc = b_mod[col];
  for (int d = 0; d < Dc; ++d) {
    float x = t[d];
    float s = x / (1.0f + __expf(-x));          // silu
    acc += s * w_mod[(size_t)d * D3 + col];
  }
  mod[(size_t)b * D3 + col] = acc;
}

// ---------------------------------------------------------------------------
// 2) xm = bf16( rmsnorm(x)*w_ln * (1+scale) + shift )      (B*L x D)
// ---------------------------------------------------------------------------
__global__ __launch_bounds__(256) void rmsnorm_mod_kernel(
    const float* __restrict__ x, const float* __restrict__ w_ln,
    const float* __restrict__ mod, unsigned short* __restrict__ xm) {
  int row = blockIdx.x;                // 0..B*L-1
  int b   = row / Lc;
  const float* xr = x + (size_t)row * Dc;
  float ss = 0.0f;
  for (int i = threadIdx.x; i < Dc; i += 256) { float v = xr[i]; ss += v * v; }
#pragma unroll
  for (int off = 16; off >= 1; off >>= 1) ss += __shfl_xor(ss, off, 32);
  __shared__ float red[8];
  __shared__ float sinv;
  if ((threadIdx.x & 31) == 0) red[threadIdx.x >> 5] = ss;
  __syncthreads();
  if (threadIdx.x == 0) {
    float t = 0.0f;
#pragma unroll
    for (int i = 0; i < 8; ++i) t += red[i];
    sinv = rsqrtf(t / (float)Dc + 1e-5f);
  }
  __syncthreads();
  float inv = sinv;
  const float* shift = mod + (size_t)b * D3;
  const float* scale = shift + Dc;
  for (int i = threadIdx.x; i < Dc; i += 256) {
    float v = xr[i] * inv * w_ln[i];
    v = v * (1.0f + scale[i]) + shift[i];
    xm[(size_t)row * Dc + i] = f2bf(v);
  }
}

// ---------------------------------------------------------------------------
// 3) WT[n][k] = bf16(W[k][n])    (weights -> transposed bf16 for B-fragments)
// ---------------------------------------------------------------------------
__global__ __launch_bounds__(256) void transpose_bf16_kernel(
    const float* __restrict__ W, unsigned short* __restrict__ WT, int K, int N) {
  size_t tid = (size_t)blockIdx.x * 256 + threadIdx.x;
  if (tid >= (size_t)K * N) return;
  int n = (int)(tid % (size_t)N);
  int k = (int)(tid / (size_t)N);
  WT[(size_t)n * K + k] = f2bf(W[tid]);
}

// ---------------------------------------------------------------------------
// 4) Tiled bf16 WMMA GEMM: C(MxN,f32) = A(MxK,bf16) * BT(NxK,bf16)^T
//    128x128 block tile, 8 waves (4x2), 32x64 wave tile, K-step 32.
//    Double-buffered LDS filled with ASYNC global->LDS copies.
//    Optional fused epilogue: C = resid + gate[b]*acc (final projection).
// ---------------------------------------------------------------------------
__global__ __launch_bounds__(256) void gemm_bf16_kernel(
    const unsigned short* __restrict__ A, const unsigned short* __restrict__ BT,
    float* __restrict__ C, int M, int N, int K,
    const float* __restrict__ resid, const float* __restrict__ gate,
    int gateStride, int rowsPerBatch) {
  __shared__ unsigned short As[2][128][40];
  __shared__ unsigned short Bs[2][128][40];
  int tid  = threadIdx.x;
  int wave = tid >> 5, lane = tid & 31;
  int ln16 = lane & 15, hb = lane >> 4;
  int wm = wave >> 1, wn = wave & 1;
  int blockM = blockIdx.y * 128, blockN = blockIdx.x * 128;

  v8f acc[2][4];
#pragma unroll
  for (int i = 0; i < 2; ++i)
#pragma unroll
    for (int j = 0; j < 4; ++j) acc[i][j] = vzero8();

  int lrow = tid >> 1;              // 0..127
  int lcol = (tid & 1) * 16;        // 0 / 16
  const unsigned short* gA = A + (size_t)(blockM + lrow) * K + lcol;
  const unsigned short* gB = BT + (size_t)(blockN + lrow) * K + lcol;

  // prologue: fill buffer 0
  async_ld_b128(&As[0][lrow][lcol],     gA);
  async_ld_b128(&As[0][lrow][lcol + 8], gA + 8);
  async_ld_b128(&Bs[0][lrow][lcol],     gB);
  async_ld_b128(&Bs[0][lrow][lcol + 8], gB + 8);

  for (int k0 = 0; k0 < K; k0 += 32) {
    int cur = (k0 >> 5) & 1;
    wait_async0();                  // this wave's fills for buf[cur] done
    __syncthreads();                // everyone's fills done
    if (k0 + 32 < K) {              // kick off next tile while we compute
      async_ld_b128(&As[cur ^ 1][lrow][lcol],     gA + k0 + 32);
      async_ld_b128(&As[cur ^ 1][lrow][lcol + 8], gA + k0 + 40);
      async_ld_b128(&Bs[cur ^ 1][lrow][lcol],     gB + k0 + 32);
      async_ld_b128(&Bs[cur ^ 1][lrow][lcol + 8], gB + k0 + 40);
    }

    Frag a[2], b[4];
#pragma unroll
    for (int i = 0; i < 2; ++i) {
      int m = wm * 32 + i * 16 + ln16;
      a[i].h[0] = *(const ushort8*)&As[cur][m][hb * 8];
      a[i].h[1] = *(const ushort8*)&As[cur][m][16 + hb * 8];
    }
#pragma unroll
    for (int j = 0; j < 4; ++j) {
      int n = wn * 64 + j * 16 + ln16;
      b[j].h[0] = *(const ushort8*)&Bs[cur][n][hb * 8];
      b[j].h[1] = *(const ushort8*)&Bs[cur][n][16 + hb * 8];
    }
#pragma unroll
    for (int i = 0; i < 2; ++i)
#pragma unroll
      for (int j = 0; j < 4; ++j) acc[i][j] = wmma_bf16(a[i], b[j], acc[i][j]);
    __syncthreads();                // done reading buf[cur]
  }

#pragma unroll
  for (int i = 0; i < 2; ++i) {
#pragma unroll
    for (int j = 0; j < 4; ++j) {
      int col = blockN + wn * 64 + j * 16 + ln16;
#pragma unroll
      for (int r = 0; r < 8; ++r) {
        int row = blockM + wm * 32 + i * 16 + hb * 8 + r;
        size_t idx = (size_t)row * N + col;
        float v = acc[i][j][r];
        if (resid) {
          int bidx = row / rowsPerBatch;
          C[idx] = resid[idx] + gate[(size_t)bidx * gateStride + col] * v;
        } else {
          C[idx] = v;
        }
      }
    }
  }
}

// ---------------------------------------------------------------------------
// 5) per-head RMSNorm + mixed RoPE -> head-major bf16 (B,H,L,Dh)
// ---------------------------------------------------------------------------
__global__ __launch_bounds__(128) void norm_rope_kernel(
    const float* __restrict__ qf, const float* __restrict__ nw,
    const float* __restrict__ cos1d, const float* __restrict__ sin1d,
    const float* __restrict__ rope3d, const long long* __restrict__ mpos,
    unsigned short* __restrict__ outr, int H) {
  int row = blockIdx.x;                 // b*L + l
  int l = row % Lc, b = row / Lc;
  int tid = threadIdx.x;                // 0..127 == dh

  bool text, img; int rel;
  {
    int in_full = 0, in_img = 0; long long relacc = 0;
    for (int s = 0; s < 2; ++s) {
      long long off = mpos[((size_t)b * 2 + s) * 2 + 0];
      long long ln  = mpos[((size_t)b * 2 + s) * 2 + 1];
      long long se  = off + (ln > 1 ? ln : 1);
      if (l >= off && l < se) in_full = 1;
      long long is = off + 1;
      if (l >= is && l < off + ln) { in_img = 1; relacc += (long long)l - is; }
    }
    text = !in_full;
    rel = (int)relacc;
    img = in_img && rel >= 0 && rel < SROPE;
    rel = rel < 0 ? 0 : (rel >= SROPE ? SROPE - 1 : rel);
  }
  float c = cos1d[(size_t)l * Dhc + tid];
  float s = sin1d[(size_t)l * Dhc + tid];
  int p = tid >> 1, j = tid & 1;
  const float* Rb = rope3d + (((size_t)rel * (Dhc / 2) + p) * 2) * 2;
  float R0 = Rb[j], R1 = Rb[2 + j];     // R[rel][p][i][j], i=0,1

  __shared__ float buf[128];
  __shared__ float wred[4];
  __shared__ float sinv;
  for (int h = 0; h < H; ++h) {
    float v = qf[(size_t)row * (H * Dhc) + h * Dhc + tid];
    float ss = v * v;
#pragma unroll
    for (int off = 16; off >= 1; off >>= 1) ss += __shfl_xor(ss, off, 32);
    if ((tid & 31) == 0) wred[tid >> 5] = ss;
    __syncthreads();
    if (tid == 0) sinv = rsqrtf((wred[0] + wred[1] + wred[2] + wred[3]) / (float)Dhc + 1e-5f);
    __syncthreads();
    float vn = v * sinv * nw[tid];
    buf[tid] = vn;
    __syncthreads();
    float partner = buf[tid ^ 64];
    float rot  = (tid < 64) ? -partner : partner;
    float vtxt = vn * c + rot * s;
    float e0 = buf[tid & ~1], e1 = buf[tid | 1];
    float vimg = e0 * R0 + e1 * R1;
    float o = text ? vtxt : (img ? vimg : vn);
    outr[(((size_t)b * H + h) * Lc + l) * Dhc + tid] = f2bf(o);
    __syncthreads();
  }
}

// ---------------------------------------------------------------------------
// 6) V: (B*L, KVH*Dh) f32 -> (B,KVH,Dh,L) bf16 (transposed for PV B-frags)
// ---------------------------------------------------------------------------
__global__ __launch_bounds__(256) void v_transpose_kernel(
    const float* __restrict__ vf, unsigned short* __restrict__ vT) {
  size_t tid = (size_t)blockIdx.x * 256 + threadIdx.x;
  size_t total = (size_t)Bc * Lc * KVHc * Dhc;
  if (tid >= total) return;
  int dh = (int)(tid & (Dhc - 1));
  size_t t2 = tid >> 7;
  int h = (int)(t2 % KVHc);
  size_t t3 = t2 / KVHc;                 // b*L + l
  int l = (int)(t3 % Lc), b = (int)(t3 / Lc);
  vT[(((size_t)b * KVHc + h) * Dhc + dh) * Lc + l] = f2bf(vf[tid]);
}

// ---------------------------------------------------------------------------
// 7) Flash attention: per block one (b,h) x 128 queries.
//    K/V 32-key tiles moved by the Tensor Data Mover (double-buffered,
//    TENSORcnt); online softmax; P staged via LDS + s_wait_dscnt;
//    16 WMMAs per key tile.
// ---------------------------------------------------------------------------
__global__ __launch_bounds__(256) void attn_kernel(
    const unsigned short* __restrict__ qr,   // (B,H,L,Dh)
    const unsigned short* __restrict__ kr,   // (B,KVH,L,Dh)
    const unsigned short* __restrict__ vt,   // (B,KVH,Dh,L)
    unsigned short* __restrict__ o) {        // (B,L,H*Dh)
  int bh = blockIdx.y;
  int b = bh / Hc, h = bh % Hc, kvh = h >> 1;
  int tid = threadIdx.x, wave = tid >> 5, lane = tid & 31;
  int ln16 = lane & 15, hb = lane >> 4;
  int qbase = blockIdx.x * 128 + wave * 16;

  __shared__ unsigned short Ks[2][32][Dhc];   // 16 KB
  __shared__ unsigned short Vs[2][Dhc][32];   // 16 KB
  __shared__ unsigned short Ps[8][16][32];    //  8 KB (per-wave P tiles)

  const unsigned short* kb = kr + (size_t)(b * KVHc + kvh) * Lc * Dhc;
  const unsigned short* vb = vt + (size_t)(b * KVHc + kvh) * Dhc * Lc;

  Frag qa[4];
  {
    const unsigned short* qrow = qr + ((size_t)bh * Lc + qbase + ln16) * Dhc;
#pragma unroll
    for (int cc = 0; cc < 4; ++cc) {
      qa[cc].h[0] = *(const ushort8*)(qrow + cc * 32 + hb * 8);
      qa[cc].h[1] = *(const ushort8*)(qrow + cc * 32 + 16 + hb * 8);
    }
  }
  v8f oacc[8];
#pragma unroll
  for (int t = 0; t < 8; ++t) oacc[t] = vzero8();
  float mrow[8], lrow[8];
#pragma unroll
  for (int r = 0; r < 8; ++r) { mrow[r] = -1e30f; lrow[r] = 0.0f; }
  const float scale = 0.08838834764831845f;   // 1/sqrt(128)

  // prologue: TDM fills buffer 0 (wave 0 only; TDM is per-wave DMA)
  if (wave == 0) {
    tdm_load_2d((unsigned)(unsigned long long)&Ks[0][0][0], kb,
                Dhc, 32, Dhc, Lc, Dhc);                 // 32 rows x 128, contig
    tdm_load_2d((unsigned)(unsigned long long)&Vs[0][0][0], vb,
                32, Dhc, Lc, Dhc, Lc);                  // 128 rows x 32, stride L
  }

  for (int kt = 0; kt < Lc; kt += 32) {
    int cur = (kt >> 5) & 1;
    if (wave == 0) {
      if (kt + 32 < Lc) {                               // prefetch next tile
        tdm_load_2d((unsigned)(unsigned long long)&Ks[cur ^ 1][0][0],
                    kb + (size_t)(kt + 32) * Dhc, Dhc, 32, Dhc, Lc, Dhc);
        tdm_load_2d((unsigned)(unsigned long long)&Vs[cur ^ 1][0][0],
                    vb + kt + 32, 32, Dhc, Lc, Dhc, Lc);
        __builtin_amdgcn_s_wait_tensorcnt(2);           // tile[cur] landed
      } else {
        __builtin_amdgcn_s_wait_tensorcnt(0);
      }
    }
    __syncthreads();

    // S = Q K^T for 32 keys -> two 16x16 tiles
    v8f s0 = vzero8(), s1 = vzero8();
#pragma unroll
    for (int cc = 0; cc < 4; ++cc) {
      Frag kb0, kb1;
      kb0.h[0] = *(const ushort8*)&Ks[cur][ln16][cc * 32 + hb * 8];
      kb0.h[1] = *(const ushort8*)&Ks[cur][ln16][cc * 32 + 16 + hb * 8];
      kb1.h[0] = *(const ushort8*)&Ks[cur][16 + ln16][cc * 32 + hb * 8];
      kb1.h[1] = *(const ushort8*)&Ks[cur][16 + ln16][cc * 32 + 16 + hb * 8];
      s0 = wmma_bf16(qa[cc], kb0, s0);
      s1 = wmma_bf16(qa[cc], kb1, s1);
    }

    // online softmax (row stats replicated per 16-lane group)
    float corr[8];
#pragma unroll
    for (int r = 0; r < 8; ++r) {
      float a0 = s0[r] * scale, a1 = s1[r] * scale;
      float mx = fmaxf(a0, a1);
#pragma unroll
      for (int off = 8; off >= 1; off >>= 1) mx = fmaxf(mx, __shfl_xor(mx, off, 32));
      float nm = fmaxf(mrow[r], mx);
      float p0 = __expf(a0 - nm), p1 = __expf(a1 - nm);
      float cf = __expf(mrow[r] - nm);
      mrow[r] = nm;
      float ps = p0 + p1;
#pragma unroll
      for (int off = 8; off >= 1; off >>= 1) ps += __shfl_xor(ps, off, 32);
      lrow[r] = lrow[r] * cf + ps;
      corr[r] = cf;
      Ps[wave][hb * 8 + r][ln16]      = f2bf(p0);
      Ps[wave][hb * 8 + r][16 + ln16] = f2bf(p1);
    }
#pragma unroll
    for (int t = 0; t < 8; ++t)
#pragma unroll
      for (int r = 0; r < 8; ++r) oacc[t][r] *= corr[r];

    // per-wave P tile written above -> wait DS, rebuild as A-fragment
    asm volatile("s_wait_dscnt 0" ::: "memory");
    Frag pa;
    pa.h[0] = *(const ushort8*)&Ps[wave][ln16][hb * 8];
    pa.h[1] = *(const ushort8*)&Ps[wave][ln16][16 + hb * 8];
#pragma unroll
    for (int t = 0; t < 8; ++t) {
      Frag vfr;
      vfr.h[0] = *(const ushort8*)&Vs[cur][t * 16 + ln16][hb * 8];
      vfr.h[1] = *(const ushort8*)&Vs[cur][t * 16 + ln16][16 + hb * 8];
      oacc[t] = wmma_bf16(pa, vfr, oacc[t]);
    }
    __syncthreads();                                    // done reading buf[cur]
  }

#pragma unroll
  for (int r = 0; r < 8; ++r) {
    float inv = 1.0f / lrow[r];
    int qrow = qbase + hb * 8 + r;
    unsigned short* orow = o + (size_t)(b * Lc + qrow) * (Hc * Dhc) + h * Dhc;
#pragma unroll
    for (int t = 0; t < 8; ++t) orow[t * 16 + ln16] = f2bf(oacc[t][r] * inv);
  }
}

// ---------------------------------------------------------------------------
// host-side orchestration
// ---------------------------------------------------------------------------
extern "C" void kernel_launch(void* const* d_in, const int* in_sizes, int n_in,
                              void* d_out, int out_size, void* d_ws, size_t ws_size,
                              hipStream_t stream) {
  const float* hidden = (const float*)d_in[0];
  const float* temb   = (const float*)d_in[1];
  const float* w_ln   = (const float*)d_in[2];
  const float* w_mod  = (const float*)d_in[3];
  const float* b_mod  = (const float*)d_in[4];
  const float* wq     = (const float*)d_in[5];
  const float* wk     = (const float*)d_in[6];
  const float* wv     = (const float*)d_in[7];
  const float* wo     = (const float*)d_in[8];
  const float* qn_w   = (const float*)d_in[9];
  const float* kn_w   = (const float*)d_in[10];
  const float* cos1d  = (const float*)d_in[11];
  const float* sin1d  = (const float*)d_in[12];
  const float* rope3d = (const float*)d_in[13];
  const long long* mpos = (const long long*)d_in[14];
  float* out = (float*)d_out;

  char* ws = (char*)d_ws;
  size_t off = 0;
  auto alloc = [&](size_t bytes) -> void* {
    off = (off + 255) & ~(size_t)255;
    void* p = ws + off;
    off += bytes;
    return p;
  };
  float* modbuf = (float*)alloc((size_t)Bc * D3 * 4);
  unsigned short* xm   = (unsigned short*)alloc((size_t)Bc * Lc * Dc * 2);
  unsigned short* wqT  = (unsigned short*)alloc((size_t)(Hc * Dhc) * Dc * 2);
  unsigned short* wkT  = (unsigned short*)alloc((size_t)(KVHc * Dhc) * Dc * 2);
  unsigned short* wvT  = (unsigned short*)alloc((size_t)(KVHc * Dhc) * Dc * 2);
  unsigned short* woT  = (unsigned short*)alloc((size_t)Dc * (Hc * Dhc) * 2);
  float* qf = (float*)alloc((size_t)Bc * Lc * Hc * Dhc * 4);
  float* kf = (float*)alloc((size_t)Bc * Lc * KVHc * Dhc * 4);
  float* vf = (float*)alloc((size_t)Bc * Lc * KVHc * Dhc * 4);
  unsigned short* qrb = (unsigned short*)alloc((size_t)Bc * Hc * Lc * Dhc * 2);
  unsigned short* krb = (unsigned short*)alloc((size_t)Bc * KVHc * Lc * Dhc * 2);
  unsigned short* vtb = (unsigned short*)alloc((size_t)Bc * KVHc * Dhc * Lc * 2);
  unsigned short* ob  = (unsigned short*)alloc((size_t)Bc * Lc * Hc * Dhc * 2);
  (void)ws_size; (void)in_sizes; (void)n_in; (void)out_size;

  const int M = Bc * Lc;                 // 4096
  const int Nq = Hc * Dhc;               // 2048
  const int Nkv = KVHc * Dhc;            // 1024

  // 1) modulation vector
  mod_kernel<<<dim3((D3 + 255) / 256, Bc), 256, 0, stream>>>(temb, w_mod, b_mod, modbuf);
  // 2) rmsnorm + scale/shift -> bf16
  rmsnorm_mod_kernel<<<M, 256, 0, stream>>>(hidden, w_ln, modbuf, xm);
  // 3) weight conversion (fp32 -> transposed bf16)
  transpose_bf16_kernel<<<(int)(((size_t)Dc * Nq + 255) / 256), 256, 0, stream>>>(wq, wqT, Dc, Nq);
  transpose_bf16_kernel<<<(int)(((size_t)Dc * Nkv + 255) / 256), 256, 0, stream>>>(wk, wkT, Dc, Nkv);
  transpose_bf16_kernel<<<(int)(((size_t)Dc * Nkv + 255) / 256), 256, 0, stream>>>(wv, wvT, Dc, Nkv);
  transpose_bf16_kernel<<<(int)(((size_t)(Hc * Dhc) * Dc + 255) / 256), 256, 0, stream>>>(wo, woT, Hc * Dhc, Dc);
  // 4) QKV projections (WMMA + async LDS copies)
  gemm_bf16_kernel<<<dim3(Nq / 128, M / 128), 256, 0, stream>>>(xm, wqT, qf, M, Nq, Dc, nullptr, nullptr, 0, Lc);
  gemm_bf16_kernel<<<dim3(Nkv / 128, M / 128), 256, 0, stream>>>(xm, wkT, kf, M, Nkv, Dc, nullptr, nullptr, 0, Lc);
  gemm_bf16_kernel<<<dim3(Nkv / 128, M / 128), 256, 0, stream>>>(xm, wvT, vf, M, Nkv, Dc, nullptr, nullptr, 0, Lc);
  // 5) qk-norm + mixed rope -> head-major bf16
  norm_rope_kernel<<<M, 128, 0, stream>>>(qf, qn_w, cos1d, sin1d, rope3d, mpos, qrb, Hc);
  norm_rope_kernel<<<M, 128, 0, stream>>>(kf, kn_w, cos1d, sin1d, rope3d, mpos, krb, KVHc);
  // 6) V transpose
  v_transpose_kernel<<<(int)(((size_t)Bc * Lc * KVHc * Dhc + 255) / 256), 256, 0, stream>>>(vf, vtb);
  // 7) flash attention (WMMA + TDM tensor loads)
  attn_kernel<<<dim3(Lc / 128, Bc * Hc), 256, 0, stream>>>(qrb, krb, vtb, ob);
  // 8) output projection + residual/gate epilogue -> fp32 out
  gemm_bf16_kernel<<<dim3(Dc / 128, M / 128), 256, 0, stream>>>(
      ob, woT, out, M, Dc, Hc * Dhc, hidden, modbuf + 2 * Dc, D3, Lc);
}
